// RUSAwareGatingNetworkWithAttention_15187004358764
// MI455X (gfx1250) — matrix operand order repaired and checked by reading, not compiled, not run on hardware
//
#include <hip/hip_runtime.h>
#include <hip/hip_bf16.h>

// ---------------------------------------------------------------------------
// Problem constants (match reference)
// ---------------------------------------------------------------------------
#define BB 8
#define MM 4
#define TT 1024
#define EE 768
#define PP 512
#define DK 64
#define DV 64
#define HH 256
#define NEXP 8
#define NTOK (BB*MM*TT)      // 32768
#define NSEQ (BB*MM)         // 32

typedef __attribute__((ext_vector_type(16))) __bf16 v16bf;
typedef __attribute__((ext_vector_type(8)))  float  v8f;

union ABreg { v16bf v; uint4 q[2]; };
union ACCreg { v8f v; float f[8]; };

static __device__ __forceinline__ unsigned short f2bf(float f) {
    union { float f; unsigned u; } v; v.f = f;
    unsigned r = v.u + 0x7FFFu + ((v.u >> 16) & 1u);   // round-to-nearest-even
    return (unsigned short)(r >> 16);
}
static __device__ __forceinline__ float bf2f(unsigned short u) {
    union { unsigned u; float f; } v; v.u = ((unsigned)u) << 16;
    return v.f;
}
static __device__ __forceinline__ float sigmoidf(float x) {
    return 1.f / (1.f + __expf(-x));
}

// ---------------------------------------------------------------------------
// CDNA5 async global->LDS copy (ASYNCcnt-tracked DMA, ISA 15.18.3 op 98).
// 16 bytes per lane: LDS byte offset in a VGPR, 64-bit global address in a
// VGPR pair. Drain with s_wait_asynccnt before the workgroup barrier.
// ---------------------------------------------------------------------------
static __device__ __forceinline__ void async_b128(const void* gsrc, void* ldst) {
    const unsigned lds = (unsigned)(uintptr_t)ldst;              // low 32 bits = LDS offset
    const unsigned long long ga = (unsigned long long)(uintptr_t)gsrc;
    asm volatile("global_load_async_to_lds_b128 %0, %1, off"
                 :: "v"(lds), "v"(ga) : "memory");
}
static __device__ __forceinline__ void wait_async0() {
    asm volatile("s_wait_asynccnt 0" ::: "memory");
}

// ---------------------------------------------------------------------------
// f32 -> bf16 conversion (weights + token embeddings)
// ---------------------------------------------------------------------------
__global__ void cvt_bf16_kernel(unsigned short* __restrict__ dst,
                                const float* __restrict__ src, int n) {
    int i = blockIdx.x * 256 + threadIdx.x;
    if (i < n) dst[i] = f2bf(src[i]);
}

// ---------------------------------------------------------------------------
// Tiled bf16 WMMA GEMM:  C[M,N] = act( A[M,K] * Bw[N,K]^T + bias )
//  - block tile 64x128, 256 threads = 8 waves; each wave owns a 16x64 strip
//    (4 WMMA tiles, A fragment reused 4x)
//  - double-buffered LDS filled by global_load_async_to_lds_b128 for chunk
//    k+1 while chunk k is consumed by WMMAs; one barrier per K-chunk
//  - optional second bf16 A source for k >= ksplit (concat [processed||h])
//  - Bw = weights in original (N x K) row-major bf16 (B tile is N-major so
//    every lane reads two 16-byte k runs per fragment)
// ---------------------------------------------------------------------------
__global__ __launch_bounds__(256)
void gemm_bf16_kernel(const unsigned short* __restrict__ A1, int lda1,
                      const unsigned short* __restrict__ A2, int lda2, int ksplit,
                      const unsigned short* __restrict__ Bw,
                      const float* __restrict__ bias,
                      unsigned short* __restrict__ Cbf, float* __restrict__ Cf,
                      int N, int K, int relu)
{
    __shared__ __align__(16) unsigned short As[2][64][40];    // 10.0 KB
    __shared__ __align__(16) unsigned short Bs[2][128][40];   // 20.0 KB

    const int tid  = threadIdx.x;
    const int bm   = blockIdx.x * 64;
    const int bn   = blockIdx.y * 128;
    const int w    = tid >> 5;
    const int L    = tid & 31;
    const int half = L >> 4;
    const int lm   = L & 15;
    const int tr   = w & 3;                    // wave M-tile (0..3)
    const int tc   = w >> 2;                   // wave N-half (0..1)
    const bool waveActive = (bn + tc * 64) < N;

    // async staging coords: A 64x32 (16B/lane), B 128x32 (2x16B/lane)
    const int arow = tid >> 2;                 // 0..63
    const int acol = (tid & 3) * 8;            // 0,8,16,24
    const int brow = tid >> 1;                 // 0..127
    const int bcol = (tid & 1) * 16;           // 0,16
    const bool bOK = (bn + brow) < N;
    const size_t aRow = (size_t)(bm + arow);

    ACCreg acc[4];
    #pragma unroll
    for (int nn = 0; nn < 4; ++nn)
        #pragma unroll
        for (int r = 0; r < 8; ++r) acc[nn].f[r] = 0.f;

    auto issue = [&](int buf, int k0) {
        const int gk = k0 + acol;
        const unsigned short* asrc = (A2 && gk >= ksplit)
            ? A2 + aRow * lda2 + (gk - ksplit)
            : A1 + aRow * lda1 + gk;
        async_b128(asrc, &As[buf][arow][acol]);
        if (bOK) {
            const unsigned short* bsrc = Bw + (size_t)(bn + brow) * K + k0 + bcol;
            async_b128(bsrc,     &Bs[buf][brow][bcol]);
            async_b128(bsrc + 8, &Bs[buf][brow][bcol + 8]);
        }
    };
    auto mma = [&](int buf) {
        if (!waveActive) return;
        ABreg a;
        const int ar = tr * 16 + lm;
        a.q[0] = *(const uint4*)&As[buf][ar][8 * half];
        a.q[1] = *(const uint4*)&As[buf][ar][16 + 8 * half];
        #pragma unroll
        for (int nn = 0; nn < 4; ++nn) {
            ABreg b;
            const int br = tc * 64 + nn * 16 + lm;
            b.q[0] = *(const uint4*)&Bs[buf][br][8 * half];
            b.q[1] = *(const uint4*)&Bs[buf][br][16 + 8 * half];
            acc[nn].v = __builtin_amdgcn_wmma_f32_16x16x32_bf16(
                false, a.v, false, b.v, (short)0, acc[nn].v, false, false);
        }
    };

    issue(0, 0);
    wait_async0();
    __syncthreads();
    int cur = 0;
    for (int k0 = 32; k0 < K; k0 += 32) {
        issue(cur ^ 1, k0);    // DMA next chunk while computing current
        mma(cur);
        wait_async0();
        __syncthreads();
        cur ^= 1;
    }
    mma(cur);

    if (waveActive) {
        const int orow = bm + tr * 16 + 8 * half;
        #pragma unroll
        for (int nn = 0; nn < 4; ++nn) {
            const int ocol = bn + tc * 64 + nn * 16 + lm;
            const float bv = bias ? bias[ocol] : 0.f;
            #pragma unroll
            for (int r = 0; r < 8; ++r) {
                float v = acc[nn].f[r] + bv;
                if (relu) v = fmaxf(v, 0.f);
                const size_t off = (size_t)(orow + r) * N + ocol;
                if (Cf)  Cf[off]  = v;
                if (Cbf) Cbf[off] = f2bf(v);
            }
        }
    }
}

// ---------------------------------------------------------------------------
// Attention over the M=4 "others": keys/values are rank-2 in (R,S), so
//   score_j = R_j*(q.Wk0) + S_j*(q.Wk1) + q.b_k   (then /8, mask j==m)
//   ctx     = a*Wv0 + b*Wv1 + b_v,  a=sum w_j R_j, b=sum w_j S_j
// ---------------------------------------------------------------------------
__global__ void attn_kernel(const float* __restrict__ q, const float* __restrict__ R,
                            const float* __restrict__ S, const float* __restrict__ Wk,
                            const float* __restrict__ bk, const float* __restrict__ Wv,
                            const float* __restrict__ bv, float* __restrict__ ctx)
{
    const int tok = blockIdx.x * 256 + threadIdx.x;
    if (tok >= NTOK) return;
    const int t = tok & (TT - 1);
    const int m = (tok >> 10) & 3;
    const int b = tok >> 12;

    const float* qr = q + (size_t)tok * DK;
    float dR = 0.f, dS = 0.f, dB = 0.f;
    #pragma unroll 8
    for (int k = 0; k < DK; ++k) {
        const float qk = qr[k];
        dR += qk * Wk[2 * k];
        dS += qk * Wk[2 * k + 1];
        dB += qk * bk[k];
    }

    float Rv[MM], Sv[MM], sc[MM];
    float mx = -3.0e38f;
    #pragma unroll
    for (int j = 0; j < MM; ++j) {
        const size_t pidx = (((size_t)(b * MM + m) * MM + j) * TT) + t;
        Rv[j] = R[pidx];
        Sv[j] = S[pidx];
        float s = (Rv[j] * dR + Sv[j] * dS + dB) * 0.125f;  // 1/sqrt(64)
        if (j == m) s = -1.0e30f;
        sc[j] = s;
        mx = fmaxf(mx, s);
    }
    float ws[MM], sum = 0.f;
    #pragma unroll
    for (int j = 0; j < MM; ++j) { ws[j] = __expf(sc[j] - mx); sum += ws[j]; }
    const float inv = 1.f / sum;
    float a = 0.f, bb2 = 0.f;
    #pragma unroll
    for (int j = 0; j < MM; ++j) { a += ws[j] * inv * Rv[j]; bb2 += ws[j] * inv * Sv[j]; }

    float* cr = ctx + (size_t)tok * DV;
    #pragma unroll 8
    for (int v = 0; v < DV; ++v)
        cr[v] = a * Wv[2 * v] + bb2 * Wv[2 * v + 1] + bv[v];
}

// ---------------------------------------------------------------------------
// GRU input projection: gi[tok][g] = b_ih[g] + U*Wih[g][0] + sum_v ctx_v*Wih[g][1+v]
// grid = (NTOK, 3), block = 256  -> g = blockIdx.y*256 + tid
// ---------------------------------------------------------------------------
__global__ void gi_kernel(const float* __restrict__ U, const float* __restrict__ ctx,
                          const float* __restrict__ Wih, const float* __restrict__ bih,
                          float* __restrict__ gi)
{
    const int tok = blockIdx.x;
    const int g   = blockIdx.y * 256 + threadIdx.x;
    __shared__ float cs[DV];
    __shared__ float uS;
    if (threadIdx.x < DV) cs[threadIdx.x] = ctx[(size_t)tok * DV + threadIdx.x];
    if (threadIdx.x == 0) uS = U[tok];
    __syncthreads();
    const float* wr = Wih + (size_t)g * (1 + DV);
    float s = bih[g] + uS * wr[0];
    #pragma unroll 8
    for (int v = 0; v < DV; ++v) s += cs[v] * wr[1 + v];
    gi[(size_t)tok * (3 * HH) + g] = s;
}

// ---------------------------------------------------------------------------
// GRU recurrence: one cooperative workgroup (32 waves), 1024 sequential steps.
// Per step: gh[32x768] = h[32x256] @ Whh^T via 96 WMMA tiles (3 per wave,
// K=256 -> 8 x wmma_bf16 each); gate math with h in registers + bf16 LDS.
// Dynamic LDS: hB (16KB) + gh (96KB) = 112KB (< 320KB WGP LDS).
// ---------------------------------------------------------------------------
__global__ __launch_bounds__(1024)
void gru_kernel(const float* __restrict__ gi, const unsigned short* __restrict__ whh,
                const float* __restrict__ bhh, unsigned short* __restrict__ h_out)
{
    extern __shared__ __align__(16) char smem[];
    unsigned short* hB = (unsigned short*)smem;                 // [32][256]
    float*          gh = (float*)(smem + NSEQ * HH * 2);        // [32][768]

    const int tid  = threadIdx.x;
    const int w    = tid >> 5;
    const int L    = tid & 31;
    const int half = L >> 4;
    const int lm   = L & 15;
    const int i     = tid & (HH - 1);   // hidden index owned by this thread
    const int nbase = tid >> 8;         // 0..3

    float hreg[8];
    #pragma unroll
    for (int u = 0; u < 8; ++u) hreg[u] = 0.f;
    #pragma unroll
    for (int u = 0; u < 8; ++u) hB[u * 1024 + tid] = 0;  // h0 = 0

    const float br = bhh[i], bz = bhh[HH + i], bn = bhh[2 * HH + i];

    for (int t = 0; t < TT; ++t) {
        __syncthreads();   // hB from previous step visible
        #pragma unroll
        for (int s = 0; s < 3; ++s) {
            const int tile = w * 3 + s;
            const int mi = tile & 1;
            const int ni = tile >> 1;
            const int m0 = mi * 16, n0 = ni * 16;
            ACCreg acc;
            #pragma unroll
            for (int r = 0; r < 8; ++r) acc.f[r] = 0.f;
            const unsigned short* arow = hB  + (size_t)(m0 + lm) * HH;
            const unsigned short* brow = whh + (size_t)(n0 + lm) * HH;
            #pragma unroll
            for (int kc = 0; kc < 8; ++kc) {
                const int k0 = kc * 32;
                ABreg a, b;
                a.q[0] = *(const uint4*)(arow + k0 + 8 * half);
                a.q[1] = *(const uint4*)(arow + k0 + 16 + 8 * half);
                b.q[0] = *(const uint4*)(brow + k0 + 8 * half);
                b.q[1] = *(const uint4*)(brow + k0 + 16 + 8 * half);
                acc.v = __builtin_amdgcn_wmma_f32_16x16x32_bf16(
                    false, a.v, false, b.v, (short)0, acc.v, false, false);
            }
            #pragma unroll
            for (int r = 0; r < 8; ++r)
                gh[(size_t)(m0 + r + 8 * half) * (3 * HH) + n0 + lm] = acc.f[r];
        }
        __syncthreads();
        #pragma unroll
        for (int u = 0; u < 8; ++u) {
            const int n = u * 4 + nbase;
            const size_t grow = ((size_t)n * TT + t) * (3 * HH);
            const float i_r = gi[grow + i];
            const float i_z = gi[grow + HH + i];
            const float i_n = gi[grow + 2 * HH + i];
            const float* ghn = gh + (size_t)n * (3 * HH);
            const float h_r = ghn[i] + br;
            const float h_z = ghn[HH + i] + bz;
            const float h_n = ghn[2 * HH + i] + bn;
            const float rr = sigmoidf(i_r + h_r);
            const float zz = sigmoidf(i_z + h_z);
            const float nn = tanhf(i_n + rr * h_n);
            const float hN = (1.f - zz) * nn + zz * hreg[u];
            hreg[u] = hN;
            const unsigned short hb = f2bf(hN);
            hB[(size_t)n * HH + i] = hb;
            h_out[((size_t)n * TT + t) * HH + i] = hb;
        }
    }
}

// ---------------------------------------------------------------------------
// Final tiny projection: logits[tok][e] = b2[e] + sum_k hid_bf[tok][k]*W2[e][k]
// ---------------------------------------------------------------------------
__global__ void mlp2_kernel(const unsigned short* __restrict__ hid,
                            const float* __restrict__ W2, const float* __restrict__ b2,
                            float* __restrict__ out)
{
    const int id = blockIdx.x * 256 + threadIdx.x;
    if (id >= NTOK * NEXP) return;
    const int tok = id >> 3;
    const int e   = id & 7;
    const unsigned short* hr = hid + (size_t)tok * 384;
    const float* wr = W2 + (size_t)e * 384;
    float s = b2[e];
    #pragma unroll 8
    for (int k = 0; k < 384; ++k) s += bf2f(hr[k]) * wr[k];
    out[id] = s;
}

// ---------------------------------------------------------------------------
// Host orchestration
// ---------------------------------------------------------------------------
extern "C" void kernel_launch(void* const* d_in, const int* in_sizes, int n_in,
                              void* d_out, int out_size, void* d_ws, size_t ws_size,
                              hipStream_t stream)
{
    const float* tok    = (const float*)d_in[0];
    const float* U      = (const float*)d_in[1];
    const float* R      = (const float*)d_in[2];
    const float* S      = (const float*)d_in[3];
    const float* W_tok  = (const float*)d_in[4];
    const float* b_tok  = (const float*)d_in[5];
    const float* W_q    = (const float*)d_in[6];
    const float* b_q    = (const float*)d_in[7];
    const float* W_k    = (const float*)d_in[8];
    const float* b_k    = (const float*)d_in[9];
    const float* W_v    = (const float*)d_in[10];
    const float* b_v    = (const float*)d_in[11];
    const float* W_ih   = (const float*)d_in[12];
    const float* W_hh   = (const float*)d_in[13];
    const float* b_ih   = (const float*)d_in[14];
    const float* b_hh   = (const float*)d_in[15];
    const float* W_mlp1 = (const float*)d_in[16];
    const float* b_mlp1 = (const float*)d_in[17];
    const float* W_mlp2 = (const float*)d_in[18];
    const float* b_mlp2 = (const float*)d_in[19];

    char* ws = (char*)d_ws;
    // workspace layout (bytes); x_bf region is later reused for gi
    unsigned short* wtok_bf  = (unsigned short*)(ws + 0);           // 512*768
    unsigned short* wq_bf    = (unsigned short*)(ws + 786432);      // 64*512
    unsigned short* wmlp1_bf = (unsigned short*)(ws + 851968);      // 384*768
    unsigned short* whh_bf   = (unsigned short*)(ws + 1441792);     // 768*256
    unsigned short* x_bf     = (unsigned short*)(ws + 2097152);     // NTOK*768 bf16 (48MB)
    float*          gi       = (float*)(ws + 2097152);              // NTOK*768 f32  (96MB, aliases x_bf)
    unsigned short* proc_bf  = (unsigned short*)(ws + 102760448);   // NTOK*512 bf16 (32MB)
    float*          qbuf     = (float*)(ws + 136314880);            // NTOK*64 f32   (8MB)
    float*          ctx      = (float*)(ws + 144703488);            // NTOK*64 f32   (8MB)
    unsigned short* h_bf     = (unsigned short*)(ws + 153092096);   // NTOK*256 bf16 (16MB)
    unsigned short* hid_bf   = (unsigned short*)(ws + 169869312);   // NTOK*384 bf16 (24MB)

    // 1) bf16 conversions (weights + token embeddings)
    cvt_bf16_kernel<<<(512 * 768 + 255) / 256, 256, 0, stream>>>(wtok_bf, W_tok, 512 * 768);
    cvt_bf16_kernel<<<(64 * 512 + 255) / 256, 256, 0, stream>>>(wq_bf, W_q, 64 * 512);
    cvt_bf16_kernel<<<(384 * 768 + 255) / 256, 256, 0, stream>>>(wmlp1_bf, W_mlp1, 384 * 768);
    cvt_bf16_kernel<<<(768 * 256 + 255) / 256, 256, 0, stream>>>(whh_bf, W_hh, 768 * 256);
    cvt_bf16_kernel<<<(NTOK * EE + 255) / 256, 256, 0, stream>>>(x_bf, tok, NTOK * EE);

    // 2) processed = relu(x @ Wtok^T + b_tok) -> bf16   (N=512, K=768)
    gemm_bf16_kernel<<<dim3(NTOK / 64, PP / 128), 256, 0, stream>>>(
        x_bf, EE, nullptr, 0, 0, wtok_bf, b_tok, proc_bf, nullptr, PP, EE, 1);

    // 3) q = processed @ Wq^T + b_q -> f32   (N=64, K=512)
    gemm_bf16_kernel<<<dim3(NTOK / 64, 1), 256, 0, stream>>>(
        proc_bf, PP, nullptr, 0, 0, wq_bf, b_q, nullptr, qbuf, DK, PP, 0);

    // 4) attention (collapsed rank-2 form) -> ctx
    attn_kernel<<<NTOK / 256, 256, 0, stream>>>(qbuf, R, S, W_k, b_k, W_v, b_v, ctx);

    // 5) GRU input projection -> gi (overwrites x_bf region; x_bf is dead now)
    gi_kernel<<<dim3(NTOK, 3), 256, 0, stream>>>(U, ctx, W_ih, b_ih, gi);

    // 6) GRU recurrence (single cooperative workgroup, WMMA per step)
    gru_kernel<<<1, 1024, NSEQ * HH * 2 + NSEQ * 3 * HH * 4, stream>>>(
        gi, whh_bf, b_hh, h_bf);

    // 7) hid = relu([processed||h] @ Wmlp1^T + b) -> bf16  (N=384, K=768, ksplit=512)
    gemm_bf16_kernel<<<dim3(NTOK / 64, 3), 256, 0, stream>>>(
        proc_bf, PP, h_bf, HH, PP, wmlp1_bf, b_mlp1, hid_bf, nullptr, 384, EE, 1);

    // 8) logits = hid @ Wmlp2^T + b2 -> d_out (f32)
    mlp2_kernel<<<(NTOK * NEXP + 255) / 256, 256, 0, stream>>>(
        hid_bf, W_mlp2, b_mlp2, (float*)d_out);
}
